// GaussianKernel_40733469836057
// MI455X (gfx1250) — matrix operand order repaired and checked by reading, not compiled, not run on hardware
//
#include <hip/hip_runtime.h>

typedef __attribute__((ext_vector_type(16))) __bf16 v16bf;
typedef __attribute__((ext_vector_type(8)))  float  v8f;
typedef __attribute__((ext_vector_type(4)))  unsigned int v4u;
typedef __attribute__((ext_vector_type(8)))  unsigned int v8u;

#define B_ROWS 16384
#define L_COLS 2048
#define F_DIM  512
#define BM 128
#define BN 128
#define BK 32
#define LDS_STRIDE 40   // 32 data + 8 pad halfwords -> 80B row stride, 16B-aligned chunks

__device__ __forceinline__ unsigned f32_bf16(float f) {
    unsigned u = __float_as_uint(f);
    u += 0x7FFFu + ((u >> 16) & 1u);   // round-to-nearest-even
    return u >> 16;
}

// ---------------------------------------------------------------------------
// Fused pass: f32 -> bf16 copy of a row-major [nrows, 512] matrix + row norms.
// One wave32 per row; coalesced float4 reads, uint2 (4xbf16) writes.
// ---------------------------------------------------------------------------
__global__ __launch_bounds__(256)
void convert_norms_kernel(const float* __restrict__ src,
                          unsigned short* __restrict__ dst,
                          float* __restrict__ norms, int nrows) {
    const int wave = threadIdx.x >> 5;
    const int lane = threadIdx.x & 31;
    const int row  = blockIdx.x * 8 + wave;
    if (row >= nrows) return;
    const float4* s4 = (const float4*)(src + (size_t)row * F_DIM);
    uint2*        d2 = (uint2*)(dst + (size_t)row * F_DIM);
    float s = 0.f;
    #pragma unroll
    for (int j = 0; j < 4; ++j) {
        float4 v = s4[lane + j * 32];
        s += v.x * v.x + v.y * v.y + v.z * v.z + v.w * v.w;
        uint2 p;
        p.x = f32_bf16(v.x) | (f32_bf16(v.y) << 16);
        p.y = f32_bf16(v.z) | (f32_bf16(v.w) << 16);
        d2[lane + j * 32] = p;
    }
    #pragma unroll
    for (int off = 16; off > 0; off >>= 1) s += __shfl_xor(s, off, 32);
    if (lane == 0) norms[row] = s;
}

// ---------------------------------------------------------------------------
// Norms only (fallback path when ws is too small for bf16 copies).
// ---------------------------------------------------------------------------
__global__ __launch_bounds__(256)
void norms_kernel(const float* __restrict__ x, const float* __restrict__ lm,
                  float* __restrict__ norms) {
    const int wave = threadIdx.x >> 5;
    const int lane = threadIdx.x & 31;
    const int row  = blockIdx.x * 8 + wave;
    if (row >= B_ROWS + L_COLS) return;
    const float* src = (row < B_ROWS) ? (x + (size_t)row * F_DIM)
                                      : (lm + (size_t)(row - B_ROWS) * F_DIM);
    float s = 0.f;
    #pragma unroll
    for (int i = 0; i < F_DIM / 32; ++i) {
        float v = src[lane + i * 32];
        s += v * v;
    }
    #pragma unroll
    for (int off = 16; off > 0; off >>= 1) s += __shfl_xor(s, off, 32);
    if (lane == 0) norms[row] = s;
}

// ---------------------------------------------------------------------------
// TDM descriptor issue: DMA one [128 rows x 32 cols] bf16 tile from a
// row-major [nrows, 512] bf16 matrix into LDS with 16B padding per 64B row
// (-> 80B LDS row stride). Wave-uniform; call from one wave only.
// D# layout per CDNA5 ISA ch.8 (group0 128b, group1 256b, groups2/3 zero).
// ---------------------------------------------------------------------------
__device__ __forceinline__ void tdm_load_tile(unsigned lds_byte_off,
                                              const unsigned short* tile_ptr,
                                              unsigned tensor_dim1) {
    unsigned long long ga = (unsigned long long)(uintptr_t)tile_ptr;
    v4u g0;
    g0.x = 1u;                                   // count=1, user mode
    g0.y = lds_byte_off;                         // lds_addr [63:32]
    g0.z = (unsigned)ga;                         // global_addr [95:64]
    g0.w = (unsigned)(ga >> 32) | (2u << 30);    // global_addr[56:32] + type=2
    v8u g1;
    g1[0] = (1u << 16)                           // data_size = 2 bytes
          | (1u << 20)                           // pad_enable
          | (3u << 22)                           // pad_interval: 16 DWORDs (64B)
          | (3u << 25);                          // pad_amount: 4 DWORDs (16B)
    g1[1] = ((unsigned)F_DIM & 0xFFFFu) << 16;   // tensor_dim0[15:0] (bits 63:48)
    g1[2] = (tensor_dim1 & 0xFFFFu) << 16;       // tensor_dim1[15:0] (bits 95:80)
    g1[3] = (tensor_dim1 >> 16) | ((unsigned)BK << 16); // dim1 hi | tile_dim0=32
    g1[4] = (unsigned)BM;                        // tile_dim1=128, tile_dim2=0
    g1[5] = (unsigned)F_DIM;                     // tensor_dim0_stride = 512
    g1[6] = 0u;
    g1[7] = 0u;
    v4u gz = {0u, 0u, 0u, 0u};
    asm volatile("tensor_load_to_lds %0, %1, %2, %3"
                 :: "s"(g0), "s"(g1), "s"(gz), "s"(gz) : "memory");
}

// ---------------------------------------------------------------------------
// Kernel 2a (TDM path): WMMA GEMM on preconverted bf16 with double-buffered
// TDM staging. Workgroup 256 threads = 8 waves; tile 128x128; wave = 2x4
// 16x16 WMMA tiles. Epilogue: d2 -> d_out, fused sum(sqrt(d2)) -> partials.
// ---------------------------------------------------------------------------
__global__ __launch_bounds__(256)
void gemm_d2_tdm(const unsigned short* __restrict__ xb,
                 const unsigned short* __restrict__ lmb,
                 const float* __restrict__ norms,
                 float* __restrict__ out, float* __restrict__ partials) {
    __shared__ unsigned short As[2][BM * LDS_STRIDE];
    __shared__ unsigned short Bs[2][BN * LDS_STRIDE];
    __shared__ float red[8];

    const int tid  = threadIdx.x;
    const int lane = tid & 31;
    const int wave = tid >> 5;
    const int wm   = wave & 3;
    const int wn   = wave >> 2;
    const int rowBase = blockIdx.y * BM;
    const int colBase = blockIdx.x * BN;

    v8f acc[2][4];
    #pragma unroll
    for (int i = 0; i < 2; ++i)
        #pragma unroll
        for (int j = 0; j < 4; ++j) {
            v8f z = {0.f, 0.f, 0.f, 0.f, 0.f, 0.f, 0.f, 0.f};
            acc[i][j] = z;
        }

    const int half = lane >> 4;
    const int l16  = lane & 15;
    const int ak0  = half * 8;      // A halves: K {0..7,16..23} / {8..15,24..31}
    const int bk0  = half * 8;      // B halves: K {0..7,16..23} / {8..15,24..31}

    union Frag { uint4 q[2]; v16bf v; };

    // prologue: DMA first tiles into buffer 0
    if (wave == 0) {
        tdm_load_tile((unsigned)(uintptr_t)&As[0][0],
                      xb + (size_t)rowBase * F_DIM, B_ROWS);
        tdm_load_tile((unsigned)(uintptr_t)&Bs[0][0],
                      lmb + (size_t)colBase * F_DIM, L_COLS);
        __builtin_amdgcn_s_wait_tensorcnt(0);
    }
    __syncthreads();

    int buf = 0;
    for (int kk = 0; kk < F_DIM; kk += BK, buf ^= 1) {
        // prefetch next k-chunk into the other buffer via TDM
        if (wave == 0 && kk + BK < F_DIM) {
            tdm_load_tile((unsigned)(uintptr_t)&As[buf ^ 1][0],
                          xb + (size_t)rowBase * F_DIM + (kk + BK), B_ROWS);
            tdm_load_tile((unsigned)(uintptr_t)&Bs[buf ^ 1][0],
                          lmb + (size_t)colBase * F_DIM + (kk + BK), L_COLS);
        }

        Frag fa[2], fb[4];
        #pragma unroll
        for (int tm = 0; tm < 2; ++tm) {
            const int m = wm * 32 + tm * 16 + l16;
            fa[tm].q[0] = *(const uint4*)(&As[buf][m * LDS_STRIDE + ak0]);
            fa[tm].q[1] = *(const uint4*)(&As[buf][m * LDS_STRIDE + ak0 + 16]);
        }
        #pragma unroll
        for (int tn = 0; tn < 4; ++tn) {
            const int n = wn * 64 + tn * 16 + l16;
            fb[tn].q[0] = *(const uint4*)(&Bs[buf][n * LDS_STRIDE + bk0]);
            fb[tn].q[1] = *(const uint4*)(&Bs[buf][n * LDS_STRIDE + bk0 + 16]);
        }
        #pragma unroll
        for (int tm = 0; tm < 2; ++tm)
            #pragma unroll
            for (int tn = 0; tn < 4; ++tn)
                acc[tm][tn] = __builtin_amdgcn_wmma_f32_16x16x32_bf16(
                    false, fa[tm].v, false, fb[tn].v,
                    (short)0, acc[tm][tn], false, false);

        if (wave == 0) __builtin_amdgcn_s_wait_tensorcnt(0);
        __syncthreads();
    }

    // epilogue: d2 = max(xsq + lsq - 2*cross, 0); store; sum sqrt(d2)
    const float* xsq = norms;
    const float* lsq = norms + B_ROWS;
    float s = 0.f;
    #pragma unroll
    for (int tm = 0; tm < 2; ++tm) {
        #pragma unroll
        for (int tn = 0; tn < 4; ++tn) {
            const int col = colBase + wn * 64 + tn * 16 + l16;
            const float lq = lsq[col];
            #pragma unroll
            for (int i = 0; i < 8; ++i) {
                const int row = rowBase + wm * 32 + tm * 16 + half * 8 + i;
                const float c  = acc[tm][tn][i];
                const float d2 = fmaxf(xsq[row] + lq - 2.0f * c, 0.0f);
                out[(size_t)row * L_COLS + col] = d2;
                s += sqrtf(d2);
            }
        }
    }
    #pragma unroll
    for (int off = 16; off > 0; off >>= 1) s += __shfl_xor(s, off, 32);
    if (lane == 0) red[wave] = s;
    __syncthreads();
    if (tid == 0) {
        float t = 0.f;
        #pragma unroll
        for (int w = 0; w < 8; ++w) t += red[w];
        partials[blockIdx.y * gridDim.x + blockIdx.x] = t;
    }
}

// ---------------------------------------------------------------------------
// Kernel 2b (fallback path): convert f32->bf16 on the fly while staging.
// ---------------------------------------------------------------------------
__global__ __launch_bounds__(256)
void gemm_d2_fallback(const float* __restrict__ x, const float* __restrict__ lm,
                      const float* __restrict__ norms,
                      float* __restrict__ out, float* __restrict__ partials) {
    __shared__ unsigned short As[BM * LDS_STRIDE];
    __shared__ unsigned short Bs[BN * LDS_STRIDE];
    __shared__ float red[8];

    const int tid  = threadIdx.x;
    const int lane = tid & 31;
    const int wave = tid >> 5;
    const int wm   = wave & 3;
    const int wn   = wave >> 2;
    const int rowBase = blockIdx.y * BM;
    const int colBase = blockIdx.x * BN;

    v8f acc[2][4];
    #pragma unroll
    for (int i = 0; i < 2; ++i)
        #pragma unroll
        for (int j = 0; j < 4; ++j) {
            v8f z = {0.f, 0.f, 0.f, 0.f, 0.f, 0.f, 0.f, 0.f};
            acc[i][j] = z;
        }

    const int sR = tid >> 3;
    const int sC = (tid & 7) * 4;
    const int half = lane >> 4;
    const int l16  = lane & 15;
    const int ak0  = half * 8;
    const int bk0  = half * 8;

    union Frag { uint4 q[2]; v16bf v; };

    for (int kk = 0; kk < F_DIM; kk += BK) {
        #pragma unroll
        for (int p = 0; p < 4; ++p) {
            const int r = sR + p * 32;
            float4 va = *(const float4*)(x  + (size_t)(rowBase + r) * F_DIM + kk + sC);
            float4 vb = *(const float4*)(lm + (size_t)(colBase + r) * F_DIM + kk + sC);
            uint2 pa, pb;
            pa.x = f32_bf16(va.x) | (f32_bf16(va.y) << 16);
            pa.y = f32_bf16(va.z) | (f32_bf16(va.w) << 16);
            pb.x = f32_bf16(vb.x) | (f32_bf16(vb.y) << 16);
            pb.y = f32_bf16(vb.z) | (f32_bf16(vb.w) << 16);
            *(uint2*)(&As[r * LDS_STRIDE + sC]) = pa;
            *(uint2*)(&Bs[r * LDS_STRIDE + sC]) = pb;
        }
        __syncthreads();

        Frag fa[2], fb[4];
        #pragma unroll
        for (int tm = 0; tm < 2; ++tm) {
            const int m = wm * 32 + tm * 16 + l16;
            fa[tm].q[0] = *(const uint4*)(&As[m * LDS_STRIDE + ak0]);
            fa[tm].q[1] = *(const uint4*)(&As[m * LDS_STRIDE + ak0 + 16]);
        }
        #pragma unroll
        for (int tn = 0; tn < 4; ++tn) {
            const int n = wn * 64 + tn * 16 + l16;
            fb[tn].q[0] = *(const uint4*)(&Bs[n * LDS_STRIDE + bk0]);
            fb[tn].q[1] = *(const uint4*)(&Bs[n * LDS_STRIDE + bk0 + 16]);
        }
        #pragma unroll
        for (int tm = 0; tm < 2; ++tm)
            #pragma unroll
            for (int tn = 0; tn < 4; ++tn)
                acc[tm][tn] = __builtin_amdgcn_wmma_f32_16x16x32_bf16(
                    false, fa[tm].v, false, fb[tn].v,
                    (short)0, acc[tm][tn], false, false);
        __syncthreads();
    }

    const float* xsq = norms;
    const float* lsq = norms + B_ROWS;
    float s = 0.f;
    #pragma unroll
    for (int tm = 0; tm < 2; ++tm) {
        #pragma unroll
        for (int tn = 0; tn < 4; ++tn) {
            const int col = colBase + wn * 64 + tn * 16 + l16;
            const float lq = lsq[col];
            #pragma unroll
            for (int i = 0; i < 8; ++i) {
                const int row = rowBase + wm * 32 + tm * 16 + half * 8 + i;
                const float c  = acc[tm][tn][i];
                const float d2 = fmaxf(xsq[row] + lq - 2.0f * c, 0.0f);
                out[(size_t)row * L_COLS + col] = d2;
                s += sqrtf(d2);
            }
        }
    }
    #pragma unroll
    for (int off = 16; off > 0; off >>= 1) s += __shfl_xor(s, off, 32);
    if (lane == 0) red[wave] = s;
    __syncthreads();
    if (tid == 0) {
        float t = 0.f;
        #pragma unroll
        for (int w = 0; w < 8; ++w) t += red[w];
        partials[blockIdx.y * gridDim.x + blockIdx.x] = t;
    }
}

// ---------------------------------------------------------------------------
// Kernel 3: deterministic reduction of partials -> gamma
// ---------------------------------------------------------------------------
__global__ __launch_bounds__(256)
void finalize_gamma(const float* __restrict__ partials, int n,
                    float* __restrict__ gamma_out) {
    __shared__ float sm[256];
    float s = 0.f;
    for (int i = threadIdx.x; i < n; i += 256) s += partials[i];
    sm[threadIdx.x] = s;
    __syncthreads();
    for (int off = 128; off > 0; off >>= 1) {
        if (threadIdx.x < off) sm[threadIdx.x] += sm[threadIdx.x + off];
        __syncthreads();
    }
    if (threadIdx.x == 0) {
        const float mean = sm[0] / ((float)B_ROWS * (float)L_COLS);
        gamma_out[0] = 1.0f / (2.0f * mean * mean);
    }
}

// ---------------------------------------------------------------------------
// Kernel 4: out = exp(-gamma * d2), in place, float4 streaming
// ---------------------------------------------------------------------------
__global__ __launch_bounds__(256)
void exp_map(float* __restrict__ out, const float* __restrict__ gamma_p) {
    const float g = gamma_p[0];
    const size_t i = ((size_t)blockIdx.x * 256 + threadIdx.x) * 4;
    float4 v = *(float4*)(out + i);
    v.x = expf(-g * v.x);
    v.y = expf(-g * v.y);
    v.z = expf(-g * v.z);
    v.w = expf(-g * v.w);
    *(float4*)(out + i) = v;
}

// ---------------------------------------------------------------------------
extern "C" void kernel_launch(void* const* d_in, const int* in_sizes, int n_in,
                              void* d_out, int out_size, void* d_ws, size_t ws_size,
                              hipStream_t stream) {
    const float* x  = (const float*)d_in[0];   // [16384, 512]
    const float* lm = (const float*)d_in[1];   // [2048, 512]
    float* out = (float*)d_out;                // [16384, 2048]

    const size_t xb_bytes  = (size_t)B_ROWS * F_DIM * 2;   // 16 MB
    const size_t lmb_bytes = (size_t)L_COLS * F_DIM * 2;   // 2 MB
    const size_t meta_floats = (size_t)(B_ROWS + L_COLS) + 2048 + 1;
    const size_t ws_needed = xb_bytes + lmb_bytes + meta_floats * sizeof(float);

    dim3 grid(L_COLS / BN, B_ROWS / BM);       // (16, 128)
    const int nparts = (int)(grid.x * grid.y); // 2048

    if (ws_size >= ws_needed) {
        // --- TDM path: preconvert to bf16, DMA-staged WMMA GEMM ---
        unsigned short* xb  = (unsigned short*)d_ws;
        unsigned short* lmb = (unsigned short*)((char*)d_ws + xb_bytes);
        float* norms    = (float*)((char*)d_ws + xb_bytes + lmb_bytes);
        float* partials = norms + (B_ROWS + L_COLS);
        float* gamma    = partials + 2048;

        hipLaunchKernelGGL(convert_norms_kernel, dim3(B_ROWS / 8), dim3(256),
                           0, stream, x, xb, norms, B_ROWS);
        hipLaunchKernelGGL(convert_norms_kernel, dim3(L_COLS / 8), dim3(256),
                           0, stream, lm, lmb, norms + B_ROWS, L_COLS);
        hipLaunchKernelGGL(gemm_d2_tdm, grid, dim3(256), 0, stream,
                           xb, lmb, norms, out, partials);
        hipLaunchKernelGGL(finalize_gamma, dim3(1), dim3(256), 0, stream,
                           partials, nparts, gamma);
        hipLaunchKernelGGL(exp_map, dim3((B_ROWS * (size_t)L_COLS) / (256 * 4)),
                           dim3(256), 0, stream, out, gamma);
    } else {
        // --- fallback: convert-on-the-fly GEMM, tiny ws footprint ---
        float* norms    = (float*)d_ws;
        float* partials = norms + (B_ROWS + L_COLS);
        float* gamma    = partials + 2048;

        hipLaunchKernelGGL(norms_kernel, dim3((B_ROWS + L_COLS) / 8), dim3(256),
                           0, stream, x, lm, norms);
        hipLaunchKernelGGL(gemm_d2_fallback, grid, dim3(256), 0, stream,
                           x, lm, norms, out, partials);
        hipLaunchKernelGGL(finalize_gamma, dim3(1), dim3(256), 0, stream,
                           partials, nparts, gamma);
        hipLaunchKernelGGL(exp_map, dim3((B_ROWS * (size_t)L_COLS) / (256 * 4)),
                           dim3(256), 0, stream, out, gamma);
    }
}
